// EdgeModel_76768245449004
// MI455X (gfx1250) — matrix operand order repaired and checked by reading, not compiled, not run on hardware
//
#include <hip/hip_runtime.h>
#include <hip/hip_bf16.h>
#include <math.h>

// ---------------------------------------------------------------------------
// Types for the CDNA5 WMMA builtins
// ---------------------------------------------------------------------------
typedef __attribute__((ext_vector_type(16))) __bf16        v16bf;
typedef __attribute__((ext_vector_type(8)))  float         v8f;
typedef __attribute__((ext_vector_type(8)))  unsigned int  v8u;
typedef __attribute__((ext_vector_type(4)))  float         v4f;   // for NT loads

// ---- float -> bf16 (round-half-up ~= RNE except exact ties).
// pack2: one v_add per input + one v_perm_b32 (bytes {hi[3],hi[2],lo[3],lo[2]}).
__device__ __forceinline__ unsigned pack2(float lo, float hi) {
    unsigned ul = __builtin_bit_cast(unsigned, lo) + 0x8000u;
    unsigned uh = __builtin_bit_cast(unsigned, hi) + 0x8000u;
    return __builtin_amdgcn_perm(uh, ul, 0x07060302u);
}
__device__ __forceinline__ __bf16 tobf(float f) {
    unsigned u = __builtin_bit_cast(unsigned, f) + 0x8000u;
    return __builtin_bit_cast(__bf16, (unsigned short)(u >> 16));
}

// Hardware tanh (gfx1250 V_TANH_F32, a TRANS32 op) — confirmed lowering.
__device__ __forceinline__ float fast_tanh(float x) {
#if defined(__has_builtin) && __has_builtin(__builtin_amdgcn_tanhf)
    return __builtin_amdgcn_tanhf(x);
#else
    float r;
    asm volatile("v_tanh_f32 %0, %1\n\tv_nop\n\tv_nop" : "=v"(r) : "v"(x));
    return r;
#endif
}

// Load a 16-bit A fragment from LDS: two 16-byte chunks per lane
// (lane half 0: K = base+0..7 and base+16..23; half 1: +8 on each)
__device__ __forceinline__ v16bf ldsA(const __bf16* p0, const __bf16* p1) {
    uint4 a = *(const uint4*)p0;
    uint4 b = *(const uint4*)p1;
    v8u t;
    t[0] = a.x; t[1] = a.y; t[2] = a.z; t[3] = a.w;
    t[4] = b.x; t[5] = b.y; t[6] = b.z; t[7] = b.w;
    return __builtin_bit_cast(v16bf, t);
}

// Load a pre-swizzled B fragment: 8 consecutive dwords per lane from global/L2
__device__ __forceinline__ v16bf ldgB(const unsigned* __restrict__ p) {
    uint4 a = *(const uint4*)p;
    uint4 b = *(const uint4*)(p + 4);
    v8u t;
    t[0] = a.x; t[1] = a.y; t[2] = a.z; t[3] = a.w;
    t[4] = b.x; t[5] = b.y; t[6] = b.z; t[7] = b.w;
    return __builtin_bit_cast(v16bf, t);
}

__device__ __forceinline__ v8f wmma_bf16(v16bf a, v16bf b, v8f c) {
    return __builtin_amdgcn_wmma_f32_16x16x32_bf16(
        /*neg_a=*/false, a, /*neg_b=*/false, b,
        /*c_mod=*/(short)0, c, /*reuse_a=*/false, /*reuse_b=*/false);
}

// ---------------------------------------------------------------------------
// Prep kernel: bf16-convert + swizzle W1/W2/W3 into WMMA B-operand layout,
// fold b2 + BatchNorm(eval) into per-column scale/shift.
// B layout per (ntile, ktile): [lane 0..31][v 0..7] u32; lane half h: v 0..3 ->
// K 8h..8h+7, v 4..7 -> K 16+8h..16+8h+7 (of the 32-wide k-tile); u32 v holds
// K=k0 (lo),k0+1 (hi) for column n*16 + lane%16.
// ---------------------------------------------------------------------------
#define W1P_N   2048              // 8 ntiles * 1 ktile  * 32 * 8
#define W2P_N   24576             // 8 ntiles * 12 ktiles * 32 * 8
#define W3P_N   8192              // 8 ntiles * 4 ktiles * 32 * 8
#define W2P_OFF (W1P_N)
#define W3P_OFF (W1P_N + W2P_N)
#define BN_OFF  (W1P_N + W2P_N + W3P_N)
#define PREP_TOTAL (BN_OFF + 128)

__global__ void prep_kernel(const float* __restrict__ W1, const float* __restrict__ W2,
                            const float* __restrict__ W3, const float* __restrict__ b2,
                            const float* __restrict__ g,  const float* __restrict__ be,
                            const float* __restrict__ mu, const float* __restrict__ var,
                            unsigned* __restrict__ W1p, unsigned* __restrict__ W2p,
                            unsigned* __restrict__ W3p, float* __restrict__ bns,
                            float* __restrict__ bnt) {
    int i = blockIdx.x * blockDim.x + threadIdx.x;
    if (i < W1P_N) {
        int v = i & 7, lane = (i >> 3) & 31, n = i >> 8;
        int k0 = ((lane >> 4) << 3) + ((v >> 2) << 4) + 2 * (v & 3);
        int c  = (n << 4) + (lane & 15);
        W1p[i] = pack2(W1[k0 * 128 + c], W1[(k0 + 1) * 128 + c]);
    } else if (i < W2P_OFF + W2P_N) {
        int j = i - W2P_OFF;
        int v = j & 7, lane = (j >> 3) & 31, kt = (j >> 8) % 12, n = j / 3072;
        int k0 = kt * 32 + ((lane >> 4) << 3) + ((v >> 2) << 4) + 2 * (v & 3);
        int c  = (n << 4) + (lane & 15);
        W2p[j] = pack2(W2[k0 * 128 + c], W2[(k0 + 1) * 128 + c]);
    } else if (i < W3P_OFF + W3P_N) {
        int j = i - W3P_OFF;
        int v = j & 7, lane = (j >> 3) & 31, kt = (j >> 8) & 3, n = j >> 10;
        int k0 = kt * 32 + ((lane >> 4) << 3) + ((v >> 2) << 4) + 2 * (v & 3);
        int c  = (n << 4) + (lane & 15);
        W3p[j] = pack2(W3[k0 * 128 + c], W3[(k0 + 1) * 128 + c]);
    } else if (i < PREP_TOTAL) {
        int c = i - BN_OFF;
        float s = g[c] * rsqrtf(var[c] + 1e-5f);
        bns[c] = s;
        bnt[c] = (b2[c] - mu[c]) * s + be[c];   // folds b2 + BN shift
    }
}

// ---------------------------------------------------------------------------
// Fused edge-model kernel: 1 block (8 waves / 256 thr) = 32 edges
// (two 16-row M-tiles; wave w -> M-tile w>>2, N-tiles (w&3) and (w&3)+4).
//   LDS A2[32][384+pad] = concat( bf16(x[row]) | bf16(x[col]) | tanh-GEMM1 )
//   GEMM1: [32,32]x[32,128]   (1 bf16 wmma k-step / wave-tile)
//   GEMM2: [32,384]x[384,128] (12 k-steps) + fused BN/ReLU -> LDS Hb
//   GEMM3: [32,128]x[128,128] (4 k-steps) + bias/ReLU -> out (fp32)
// Cache policy: x-table gathers use default (RT) policy -> stays L2-resident
// (51 MB vs 192 MB L2); attr loads and output stores are non-temporal streams.
// ---------------------------------------------------------------------------
#define EPB 32                    // edges per block

__global__ __launch_bounds__(256) void edge_fused_kernel(
    const float* __restrict__ x, const long long* __restrict__ ei,
    const float* __restrict__ attr, const float* __restrict__ b1,
    const float* __restrict__ b3,
    const unsigned* __restrict__ W1p, const unsigned* __restrict__ W2p,
    const unsigned* __restrict__ W3p,
    const float* __restrict__ bns, const float* __restrict__ bnt,
    float* __restrict__ out, int E)
{
    __shared__ __bf16 A2[EPB][392];   // 384 + 8 pad (bank skew); rows 16B aligned
    __shared__ __bf16 At[EPB][32];    // edge_attr tile
    __shared__ __bf16 Hb[EPB][136];   // hidden tile, 128 + 8 pad
    __shared__ int    rowi[EPB], coli[EPB];

    const int tid = threadIdx.x;
    const int e0  = blockIdx.x * EPB;

    if (tid < EPB) {
        int e = e0 + tid;
        long long r = 0, c = 0;
        if (e < E) { r = ei[e]; c = ei[(long long)E + e]; }
        rowi[tid] = (int)r;
        coli[tid] = (int)c;
    }
    __syncthreads();

    // ---- gather x[row] | x[col] -> A2 cols [0,256), bf16 (L2-resident table)
    for (int s = tid; s < EPB * 64; s += 256) {  // EPB*256 elems / 4 per slot
        int flat = s << 2;
        int m = flat >> 8;                       // edge in tile
        int c = flat & 255;                      // concat column
        int srcRow = (c < 128) ? rowi[m] : coli[m];
        int cc = c & 127;
        const float4 v = *(const float4*)(x + (long long)srcRow * 128 + cc);
        *(uint2*)&A2[m][c] = make_uint2(pack2(v.x, v.y), pack2(v.z, v.w));
    }
    // ---- edge_attr tile -> At (bf16): EPB*32 = 1024 elems = 256 float4 slots
    {
        int flat = tid << 2;
        int m = flat >> 5;
        int c = flat & 31;
        v4f v = {0.f, 0.f, 0.f, 0.f};
        if (e0 + m < E)
            v = __builtin_nontemporal_load(
                    (const v4f*)(attr + (long long)(e0 + m) * 32 + c));
        *(uint2*)&At[m][c] = make_uint2(pack2(v.x, v.y), pack2(v.z, v.w));
    }
    __syncthreads();

    const int w    = tid >> 5;
    const int lane = tid & 31;
    const int half = lane >> 4;
    const int lm   = lane & 15;
    const int mt   = w >> 2;                     // M-tile (0/1)
    const int mrow = mt * 16 + lm;               // this lane's A row
    const int n0 = (w & 3), n1 = (w & 3) + 4;    // two N-tiles per wave

    // ---- GEMM1 + tanh  -> A2 cols [256,384)
    #pragma unroll
    for (int t = 0; t < 2; ++t) {
        int n = (t == 0) ? n0 : n1;
        v16bf a = ldsA(&At[mrow][half * 8], &At[mrow][16 + half * 8]);
        v16bf b = ldgB(W1p + (unsigned)(n * 32 + lane) * 8);
        v8f acc = {};
        acc = wmma_bf16(a, b, acc);
        int c = n * 16 + lm;
        float bias = b1[c];
        #pragma unroll
        for (int r = 0; r < 8; ++r) {
            int m = mt * 16 + r + half * 8;
            A2[m][256 + c] = tobf(fast_tanh(acc[r] + bias));
        }
    }
    __syncthreads();

    // ---- GEMM2 over K=384 (12 k-steps), fused b2+BN+ReLU -> Hb
    v8f acc0 = {}, acc1 = {};
    #pragma unroll
    for (int kt = 0; kt < 12; ++kt) {
        v16bf a  = ldsA(&A2[mrow][kt * 32 + half * 8],
                        &A2[mrow][kt * 32 + 16 + half * 8]);
        v16bf bA = ldgB(W2p + (unsigned)((n0 * 12 + kt) * 32 + lane) * 8);
        v16bf bB = ldgB(W2p + (unsigned)((n1 * 12 + kt) * 32 + lane) * 8);
        acc0 = wmma_bf16(a, bA, acc0);
        acc1 = wmma_bf16(a, bB, acc1);
    }
    {
        int cA = n0 * 16 + lm, cB = n1 * 16 + lm;
        float sA = bns[cA], tA = bnt[cA];
        float sB = bns[cB], tB = bnt[cB];
        #pragma unroll
        for (int r = 0; r < 8; ++r) {
            int m = mt * 16 + r + half * 8;
            Hb[m][cA] = tobf(fmaxf(fmaf(acc0[r], sA, tA), 0.f));
            Hb[m][cB] = tobf(fmaxf(fmaf(acc1[r], sB, tB), 0.f));
        }
    }
    __syncthreads();

    // ---- GEMM3 over K=128 (4 k-steps), bias + ReLU -> out (fp32, NT stream)
    v8f o0 = {}, o1 = {};
    #pragma unroll
    for (int kt = 0; kt < 4; ++kt) {
        v16bf a  = ldsA(&Hb[mrow][kt * 32 + half * 8],
                        &Hb[mrow][kt * 32 + 16 + half * 8]);
        v16bf bA = ldgB(W3p + (unsigned)((n0 * 4 + kt) * 32 + lane) * 8);
        v16bf bB = ldgB(W3p + (unsigned)((n1 * 4 + kt) * 32 + lane) * 8);
        o0 = wmma_bf16(a, bA, o0);
        o1 = wmma_bf16(a, bB, o1);
    }
    {
        int cA = n0 * 16 + lm, cB = n1 * 16 + lm;
        float bA = b3[cA], bB = b3[cB];
        #pragma unroll
        for (int r = 0; r < 8; ++r) {
            int m = mt * 16 + r + half * 8;
            if (e0 + m < E) {
                __builtin_nontemporal_store(fmaxf(o0[r] + bA, 0.f),
                                            out + (long long)(e0 + m) * 128 + cA);
                __builtin_nontemporal_store(fmaxf(o1[r] + bB, 0.f),
                                            out + (long long)(e0 + m) * 128 + cB);
            }
        }
    }
}

// ---------------------------------------------------------------------------
// Launcher
// ---------------------------------------------------------------------------
extern "C" void kernel_launch(void* const* d_in, const int* in_sizes, int n_in,
                              void* d_out, int out_size, void* d_ws, size_t ws_size,
                              hipStream_t stream) {
    const float*     x    = (const float*)d_in[0];
    const long long* ei   = (const long long*)d_in[1];
    const float*     attr = (const float*)d_in[2];
    const float*     W1   = (const float*)d_in[3];
    const float*     b1   = (const float*)d_in[4];
    const float*     W2   = (const float*)d_in[5];
    const float*     b2   = (const float*)d_in[6];
    const float*     g    = (const float*)d_in[7];
    const float*     be   = (const float*)d_in[8];
    const float*     mu   = (const float*)d_in[9];
    const float*     var  = (const float*)d_in[10];
    const float*     W3   = (const float*)d_in[11];
    const float*     b3   = (const float*)d_in[12];
    float*           out  = (float*)d_out;

    const int E = in_sizes[1] / 2;               // edge_index is [2, E]

    unsigned* W1p = (unsigned*)d_ws;
    unsigned* W2p = W1p + W2P_OFF;
    unsigned* W3p = W1p + W3P_OFF;
    float*    bns = (float*)(W1p + BN_OFF);
    float*    bnt = bns + 128;

    // 1) weight bf16-swizzle + BN fold (tiny, ~35K threads)
    prep_kernel<<<(PREP_TOTAL + 255) / 256, 256, 0, stream>>>(
        W1, W2, W3, b2, g, be, mu, var, W1p, W2p, W3p, bns, bnt);

    // 2) fused edge model: 32 edges per 256-thread block
    int nblk = (E + EPB - 1) / EPB;
    edge_fused_kernel<<<nblk, 256, 0, stream>>>(
        x, ei, attr, b1, b3, W1p, W2p, W3p, bns, bnt, out, E);
}
